// ConjunctionLayer_36301063585946
// MI455X (gfx1250) — matrix operand order repaired and checked by reading, not compiled, not run on hardware
//
#include <hip/hip_runtime.h>

// ConjunctionLayer on MI455X (gfx1250).
// Replace 268M v_log_f32 (transcendental-bound, ~quarter-rate) with a
// truncated Mercator series  log(1-t) = -sum_k t^k/k  (valid: t = (1-x)*w
// in [0, 0.5] given x in [0,1], W in [0,0.5]):
//   s[b,j] = -sum_{k=1..16} (1/k) * (U^(k) @ (W^(k))^T)[b,j],  U = 1-x
//   out    = -1/(-1+s) = 1/(1+acc)
// Every term is a 4096x128x512 GEMM -> v_wmma_f32_16x16x32_f16.
// Accuracy: k=1 term is fp16-compensated (uh*wh + uh*wl + ul*wh) ~ fp32;
// k>=2 terms decay as 0.5^k; Taylor tail at K=16 <= 2*0.5^17/17 ~ 9e-7
// absolute on s -> ~1e-8 relative on the output.
// Occupancy: the i-reduction is linear, so D is split across blockIdx.y into
// independent slices writing fp32 partials to d_ws (pure writes, no init
// needed); a small combine kernel sums slices and applies 1/(1+s). 4-way
// split -> 256 blocks x 8 waves = 2048 waves, ~2 waves/SIMD on a ~1024-SIMD
// part, vs 512 waves single-pass. Falls back deterministically if d_ws is
// too small.

typedef __attribute__((ext_vector_type(16))) _Float16 v16h;
typedef __attribute__((ext_vector_type(8)))  _Float16 v8h;
typedef __attribute__((ext_vector_type(8)))  float    v8f;

namespace {
constexpr int kB       = 4096;   // batch rows
constexpr int kD       = 512;    // inner dim
constexpr int kN       = 128;    // output cols
constexpr int kMBlk    = 64;     // x-rows per workgroup
constexpr int kDC      = 32;     // K-chunk == one WMMA K
constexpr int kChunks  = kD / kDC;  // 16
constexpr int kThreads = 256;    // 8 waves (wave32)
constexpr int kSeries  = 16;     // Mercator truncation order
constexpr int kCTiles  = 4;      // 16-wide column tiles per wave (4*16=64 cols)
}

// A-fragment (16x32 f16), ISA 7.12.2: lanes 0-15 hold M=lane with
// K = 8*hi + e for e<8 (VGPR0..3) and K = 16 + 8*hi + (e-8) for e>=8.
__device__ __forceinline__ v16h load_a_frag(const _Float16* p, int row, int hi) {
  const _Float16* q = p + row * kDC + hi * 8;
  v8h lo = *(const v8h*)q;
  v8h hh = *(const v8h*)(q + 16);
  v16h a;
#pragma unroll
  for (int e = 0; e < 8; ++e) { a[e] = lo[e]; a[e + 8] = hh[e]; }
  return a;
}

// B-fragment (32x16 f16): lane n = lane&15 is the column; element e maps to
// K = 8*hi + e for e<8 and K = 16 + 8*hi + (e-8) for e>=8 (same as A).
__device__ __forceinline__ v16h load_b_frag(const _Float16* p, int j, int hi) {
  const _Float16* q = p + j * kDC + hi * 8;
  v8h lo = *(const v8h*)q;
  v8h hh = *(const v8h*)(q + 16);
  v16h b;
#pragma unroll
  for (int e = 0; e < 8; ++e) { b[e] = lo[e]; b[e + 8] = hh[e]; }
  return b;
}

__global__ __launch_bounds__(kThreads)
void conj_series_wmma(const float* __restrict__ X, const float* __restrict__ Wm,
                      float* __restrict__ dst, int chunks_per_slice, int do_final) {
  __shared__ __attribute__((aligned(32))) _Float16 s_u_hi[kMBlk * kDC];  //  4 KB
  __shared__ __attribute__((aligned(32))) _Float16 s_u_lo[kMBlk * kDC];  //  4 KB
  __shared__ __attribute__((aligned(32))) _Float16 s_w_hi[kN * kDC];     //  8 KB
  __shared__ __attribute__((aligned(32))) _Float16 s_w_lo[kN * kDC];     //  8 KB
  __shared__ __attribute__((aligned(32))) _Float16 s_w_pw[2][kN * kDC];  // 16 KB ping-pong

  const int tid  = threadIdx.x;
  const int lane = tid & 31;
  const int wave = tid >> 5;
  const int hi   = lane >> 4;
  const int nloc = lane & 15;
  const int rg   = wave & 3;       // row group: rows rg*16 .. rg*16+15
  const int ch   = wave >> 2;      // column half: cols ch*64 .. ch*64+63
  const int row0 = blockIdx.x * kMBlk;
  const int arow = rg * 16 + nloc; // this wave's A row inside the LDS tile
  const int c0   = blockIdx.y * chunks_per_slice;   // this block's D-slice
  const int c1   = c0 + chunks_per_slice;

  v8f acc[kCTiles] = {};           // 4 x (16x16 f32) output tiles per wave

#pragma clang loop unroll(disable)
  for (int dc = c0; dc < c1; ++dc) {
    const int dbase = dc * kDC;

    if (dc + 1 < c1) {             // warm L2/WGP$ for the next K-chunk
      __builtin_prefetch(&X[(row0 + (tid & (kMBlk - 1))) * kD + dbase + kDC], 0, 1);
      __builtin_prefetch(&Wm[(tid & (kN - 1)) * kD + dbase + kDC], 0, 1);
    }

    // Stage u = 1-x and w into LDS, each split into fp16 hi/lo.
    for (int idx = tid; idx < kMBlk * kDC; idx += kThreads) {
      int r = idx / kDC, c = idx % kDC;
      float u = 1.0f - X[(row0 + r) * kD + dbase + c];
      _Float16 uh = (_Float16)u;
      s_u_hi[idx] = uh;
      s_u_lo[idx] = (_Float16)(u - (float)uh);
    }
    for (int idx = tid; idx < kN * kDC; idx += kThreads) {
      int j = idx / kDC, c = idx % kDC;
      float w = Wm[j * kD + dbase + c];
      _Float16 wh = (_Float16)w;
      s_w_hi[idx] = wh;
      s_w_lo[idx] = (_Float16)(w - (float)wh);
    }
    __syncthreads();

    // A fragments for this chunk; A-side powers iterate in registers.
    v16h a_hi = load_a_frag(s_u_hi, arow, hi);
    v16h a_lo = load_a_frag(s_u_lo, arow, hi);
    v16h a_pw = a_hi;

    // ---- k = 1, compensated: u*w ~= uh*wh + uh*wl + ul*wh ----
#pragma unroll
    for (int ct = 0; ct < kCTiles; ++ct) {
      const int j0 = ch * 64 + ct * 16 + nloc;
      v16h b_h = load_b_frag(s_w_hi, j0, hi);
      v16h b_l = load_b_frag(s_w_lo, j0, hi);
      acc[ct] = __builtin_amdgcn_wmma_f32_16x16x32_f16(
          false, a_hi, false, b_h, (short)0, acc[ct], false, false);
      acc[ct] = __builtin_amdgcn_wmma_f32_16x16x32_f16(
          false, a_hi, false, b_l, (short)0, acc[ct], false, false);
      acc[ct] = __builtin_amdgcn_wmma_f32_16x16x32_f16(
          false, a_lo, false, b_h, (short)0, acc[ct], false, false);
    }

    // ---- k = 2 .. kSeries: acc += (u^k/k) (w^k)^T ----
#pragma clang loop unroll(disable)
    for (int k = 2; k <= kSeries; ++k) {
      // Shared B-side power update into the ping-pong buffer (k&1).
      const _Float16* wsrc = (k == 2) ? s_w_hi : s_w_pw[(k - 1) & 1];
      _Float16*       wdst = s_w_pw[k & 1];
      for (int idx = tid; idx < kN * kDC; idx += kThreads)
        wdst[idx] = (_Float16)(wsrc[idx] * s_w_hi[idx]);

      // Register-side A power update; fold 1/k into A (8 pk-muls, not per tile).
#pragma unroll
      for (int e = 0; e < 16; ++e) a_pw[e] = (_Float16)(a_pw[e] * a_hi[e]);
      const _Float16 invk = (_Float16)(1.0f / (float)k);
      v16h a_s;
#pragma unroll
      for (int e = 0; e < 16; ++e) a_s[e] = (_Float16)(a_pw[e] * invk);

      __syncthreads();   // wdst ready; ping-pong removes the WAR barrier

#pragma unroll
      for (int ct = 0; ct < kCTiles; ++ct) {
        v16h b = load_b_frag(wdst, ch * 64 + ct * 16 + nloc, hi);
        acc[ct] = __builtin_amdgcn_wmma_f32_16x16x32_f16(
            false, a_s, false, b, (short)0, acc[ct], false, false);
      }
    }
    __syncthreads();   // all reads of w_hi/w_lo done before restaging
  }

  // acc holds the partial -s for this D-slice.
  // C/D f32 16x16 layout: lane gives col n = lane&15; VGPR r gives row
  // m = r + 8*(lane>>4).
  float* base = dst + (size_t)blockIdx.y * kB * kN;
#pragma unroll
  for (int ct = 0; ct < kCTiles; ++ct) {
#pragma unroll
    for (int r = 0; r < 8; ++r) {
      int m = rg * 16 + r + 8 * hi;
      int j = ch * 64 + ct * 16 + nloc;
      float v = acc[ct][r];
      base[(row0 + m) * kN + j] = do_final ? (1.0f / (1.0f + v)) : v;
    }
  }
}

// Sum D-slice partials and apply out = -1/(-1+s) = 1/(1+acc). ~10 MB of
// traffic at 23.3 TB/s: well under half a microsecond.
__global__ __launch_bounds__(256)
void conj_combine(const float* __restrict__ part, float* __restrict__ out,
                  int nslice) {
  int i = blockIdx.x * 256 + threadIdx.x;
  if (i < kB * kN) {
    float s = 0.0f;
    for (int p = 0; p < nslice; ++p) s += part[(size_t)p * kB * kN + i];
    out[i] = 1.0f / (1.0f + s);
  }
}

extern "C" void kernel_launch(void* const* d_in, const int* in_sizes, int n_in,
                              void* d_out, int out_size, void* d_ws, size_t ws_size,
                              hipStream_t stream) {
  (void)in_sizes; (void)n_in; (void)out_size;
  const float* x  = (const float*)d_in[0];   // (4096, 512) fp32
  const float* Wm = (const float*)d_in[1];   // (128, 512)  fp32
  float* out = (float*)d_out;                // (4096, 128) fp32

  const size_t sliceBytes = (size_t)kB * kN * sizeof(float);
  int nslice = 1;                            // deterministic for fixed ws_size
  if (d_ws && ws_size >= 4 * sliceBytes)      nslice = 4;
  else if (d_ws && ws_size >= 2 * sliceBytes) nslice = 2;

  if (nslice > 1) {
    dim3 grid(kB / kMBlk, nslice);           // 128 or 256 workgroups
    hipLaunchKernelGGL(conj_series_wmma, grid, dim3(kThreads), 0, stream,
                       x, Wm, (float*)d_ws, kChunks / nslice, 0);
    hipLaunchKernelGGL(conj_combine, dim3((kB * kN + 255) / 256), dim3(256),
                       0, stream, (const float*)d_ws, out, nslice);
  } else {
    dim3 grid(kB / kMBlk, 1);                // single-pass fallback
    hipLaunchKernelGGL(conj_series_wmma, grid, dim3(kThreads), 0, stream,
                       x, Wm, out, kChunks, 1);
  }
}
